// MHGCN_33500744909075
// MI455X (gfx1250) — compile-verified, matrix operand order checked by reading, not run on hardware
//
#include <hip/hip_runtime.h>
#include <math.h>

#define N 4096
#define NFEAT 128
#define OUT 64

typedef __attribute__((ext_vector_type(8)))  _Float16 v8h;
typedef __attribute__((ext_vector_type(16))) _Float16 v16h;
typedef __attribute__((ext_vector_type(8)))  float    v8f;

// ---------------------------------------------------------------------------
// Kernel 1: build final_A = merged + merged^T + s*tanh(0.6*mult + 0.4*cond)
// as f16.  64x64 tiles; transposed merged tile staged through LDS so both
// global read patterns are coalesced.
// ---------------------------------------------------------------------------
__global__ void build_final(const float* __restrict__ A,      // 7 x N x N
                            const float* __restrict__ wb,     // 7
                            const float* __restrict__ Mri,    // 3x3
                            const float* __restrict__ strength,
                            _Float16* __restrict__ F) {
  __shared__ float Tm[64][65];  // pitch 65 -> conflict-free transposed reads
  const int n0 = blockIdx.y * 64, m0 = blockIdx.x * 64;
  const size_t NN = (size_t)N * N;

  float w[7];
#pragma unroll
  for (int r = 0; r < 7; ++r) w[r] = wb[r];
  const float M01 = Mri[1], M02 = Mri[2], M10 = Mri[3];
  const float M12 = Mri[5], M20 = Mri[6], M21 = Mri[7];  // diag removed by (1-eye)
  const float s = strength[0];

  // phase 1: Tm[i][j] = merged[m0+i][n0+j]   (coalesced along j)
  for (int e = threadIdx.x; e < 64 * 64; e += blockDim.x) {
    const int i = e >> 6, j = e & 63;
    const size_t idx = (size_t)(m0 + i) * N + (n0 + j);
    float acc = 0.f;
#pragma unroll
    for (int r = 0; r < 7; ++r) acc += w[r] * A[r * NN + idx];
    Tm[i][j] = acc;
  }
  __syncthreads();

  // phase 2: direct reads at (n0+i, m0+j); r=0..2 reused for interaction term
  for (int e = threadIdx.x; e < 64 * 64; e += blockDim.x) {
    const int i = e >> 6, j = e & 63;
    const size_t idx = (size_t)(n0 + i) * N + (m0 + j);
    float a[7];
#pragma unroll
    for (int r = 0; r < 7; ++r) a[r] = A[r * NN + idx];
    float merged = 0.f;
#pragma unroll
    for (int r = 0; r < 7; ++r) merged += w[r] * a[r];
    const float r0 = a[0], r1 = a[1], r2 = a[2];
    const float mult = (M01 + M10) * r0 * r1 + (M02 + M20) * r0 * r2 +
                       (M12 + M21) * r1 * r2;
    const float c0 = (r0 > 0.f) ? 1.f : 0.f;
    const float c1 = (r1 > 0.f) ? 1.f : 0.f;
    const float c2 = (r2 > 0.f) ? 1.f : 0.f;
    const float cond = M01 * c0 * r1 + M02 * c0 * r2 + M10 * c1 * r0 +
                       M12 * c1 * r2 + M20 * c2 * r0 + M21 * c2 * r1;
    const float inter = tanhf(0.6f * mult + 0.4f * cond);
    F[idx] = (_Float16)(merged + Tm[j][i] + s * inter);
  }
}

// ---------------------------------------------------------------------------
// Small GEMM: Y(fp32 Nx64, optional) / YT(f16 64xN, optional) = X(NxK) @ W(Kx64)
// ---------------------------------------------------------------------------
__global__ void xw_small(const float* __restrict__ X, const float* __restrict__ W,
                         int K, float* __restrict__ Y, _Float16* __restrict__ YT) {
  __shared__ float Ws[NFEAT * OUT];  // 32 KB max
  for (int e = threadIdx.x; e < K * OUT; e += blockDim.x) Ws[e] = W[e];
  __syncthreads();
  const int idx = blockIdx.x * blockDim.x + threadIdx.x;
  const int n = idx >> 6, o = idx & 63;
  const float* xr = X + (size_t)n * K;
  float acc = 0.f;
  for (int k = 0; k < K; ++k) acc += xr[k] * Ws[k * OUT + o];
  if (Y)  Y[(size_t)n * OUT + o] = acc;
  if (YT) YT[(size_t)o * N + n] = (_Float16)acc;
}

// ---------------------------------------------------------------------------
// WMMA fragment load (ISA 16-bit A-matrix 16x32 layout):
// lanes 0-15 hold K {k0..k0+7, k0+16..k0+23}; lanes 16-31 hold {+8} offsets.
// Two contiguous b128 loads per lane.
// ---------------------------------------------------------------------------
__device__ __forceinline__ v16h load_frag16(const _Float16* __restrict__ p) {
  const v8h lo = *(const v8h*)p;
  const v8h up = *(const v8h*)(p + 16);
  v16h f;
#pragma unroll
  for (int i = 0; i < 8; ++i) { f[i] = lo[i]; f[i + 8] = up[i]; }
  return f;
}

// ---------------------------------------------------------------------------
// Big GEMM: U(Nx64 fp32) = Afinal(NxN f16) @ BT^T + bias.
// 32 blocks x 8 waves; each wave -> 16 rows x 64 cols (4 WMMA tiles).
// sched_barrier(0) pins all 10 fragment loads of a K-step BEFORE the 4
// WMMAs: one load clause + one wait + 4 back-to-back v_wmma, instead of
// the scheduler's load/wait/wmma interleave (4x exposed latency).
// ---------------------------------------------------------------------------
__global__ void __launch_bounds__(256, 1)
gemm_wmma(const _Float16* __restrict__ Af,
          const _Float16* __restrict__ BT,   // 64 x N (f16)
          const float* __restrict__ bias,
          float* __restrict__ U) {
  const int lane = threadIdx.x & 31;
  const int wave = threadIdx.x >> 5;
  const int lanelo = lane & 15;
  const int hi = lane >> 4;
  const int rowBase = blockIdx.x * 128 + wave * 16;
  const int arow = rowBase + lanelo;
  const int koff = hi ? 8 : 0;

  // per-lane base pointers (k advances by +32 halves each step)
  const _Float16* ap = Af + (size_t)arow * N + koff;
  const _Float16* bp0 = BT + (size_t)(0 * 16 + lanelo) * N + koff;
  const _Float16* bp1 = BT + (size_t)(1 * 16 + lanelo) * N + koff;
  const _Float16* bp2 = BT + (size_t)(2 * 16 + lanelo) * N + koff;
  const _Float16* bp3 = BT + (size_t)(3 * 16 + lanelo) * N + koff;

  v8f acc[4];
#pragma unroll
  for (int t = 0; t < 4; ++t) {
    const float bv = bias[t * 16 + lanelo];
#pragma unroll
    for (int v = 0; v < 8; ++v) acc[t][v] = bv;
  }

  for (int k0 = 0; k0 < N; k0 += 32) {
    if (k0 + 256 < N)
      __builtin_prefetch(ap + k0 + 256, 0, 3);  // -> global_prefetch_b8

    // one clause of 10 b128 loads ...
    const v16h af = load_frag16(ap + k0);
    const v16h b0 = load_frag16(bp0 + k0);
    const v16h b1 = load_frag16(bp1 + k0);
    const v16h b2 = load_frag16(bp2 + k0);
    const v16h b3 = load_frag16(bp3 + k0);

    // ... scheduling fence (no ISA emitted): loads may not sink below,
    // WMMAs may not hoist above ...
    __builtin_amdgcn_sched_barrier(0);

    // ... then 4 back-to-back WMMAs
    acc[0] = __builtin_amdgcn_wmma_f32_16x16x32_f16(
        false, af, false, b0, (short)0, acc[0], false, false);
    acc[1] = __builtin_amdgcn_wmma_f32_16x16x32_f16(
        false, af, false, b1, (short)0, acc[1], false, false);
    acc[2] = __builtin_amdgcn_wmma_f32_16x16x32_f16(
        false, af, false, b2, (short)0, acc[2], false, false);
    acc[3] = __builtin_amdgcn_wmma_f32_16x16x32_f16(
        false, af, false, b3, (short)0, acc[3], false, false);
  }

  // C/D layout: VGPR v -> M = v (lanes 0-15) or v+8 (lanes 16-31); N = lanelo
#pragma unroll
  for (int t = 0; t < 4; ++t)
#pragma unroll
    for (int v = 0; v < 8; ++v)
      U[(size_t)(rowBase + hi * 8 + v) * OUT + t * 16 + lanelo] = acc[t][v];
}

// ---------------------------------------------------------------------------
// Rank-7 struct path: S[r][o] = sum_n encode[n][r] * X[n][o]
// ---------------------------------------------------------------------------
__global__ void etx(const float* __restrict__ E, const float* __restrict__ X,
                    float* __restrict__ S) {
  const int r = blockIdx.x >> 6, o = blockIdx.x & 63;
  float acc = 0.f;
  for (int n = threadIdx.x; n < N; n += 256)
    acc += E[n * 7 + r] * X[(size_t)n * OUT + o];
  __shared__ float red[256];
  red[threadIdx.x] = acc;
  __syncthreads();
  for (int s = 128; s > 0; s >>= 1) {
    if (threadIdx.x < s) red[threadIdx.x] += red[threadIdx.x + s];
    __syncthreads();
  }
  if (threadIdx.x == 0) S[blockIdx.x] = red[0];
}

// U[n][o] = bias[o] + sum_r encode[n][r]*sw[r]*S[r][o]
__global__ void exs(const float* __restrict__ E, const float* __restrict__ sw,
                    const float* __restrict__ S, const float* __restrict__ bias,
                    float* __restrict__ U) {
  __shared__ float Ss[7 * OUT];
  __shared__ float sws[7];
  for (int e = threadIdx.x; e < 7 * OUT; e += blockDim.x) Ss[e] = S[e];
  if (threadIdx.x < 7) sws[threadIdx.x] = sw[threadIdx.x];
  __syncthreads();
  const int idx = blockIdx.x * blockDim.x + threadIdx.x;
  const int n = idx >> 6, o = idx & 63;
  float acc = bias[o];
#pragma unroll
  for (int r = 0; r < 7; ++r) acc += E[n * 7 + r] * sws[r] * Ss[r * OUT + o];
  U[(size_t)n * OUT + o] = acc;
}

// ---------------------------------------------------------------------------
// Fused combine + three row-wise L2 norms (tuple output, concatenated)
// ---------------------------------------------------------------------------
__global__ void finalize(const float* __restrict__ U1, const float* __restrict__ U2,
                         const float* __restrict__ U4, float* __restrict__ out) {
  const int n = blockIdx.x, o = threadIdx.x;  // 64 threads
  const size_t i = (size_t)n * OUT + o;
  const float u1 = U1[i], u2 = U2[i], u4 = U4[i];
  const float br1 = 0.5f * (u1 + u2);
  const float res = 0.5f * (br1 + u4);
  __shared__ float red[3][64];
  red[0][o] = res * res; red[1][o] = br1 * br1; red[2][o] = u4 * u4;
  __syncthreads();
  for (int s = 32; s > 0; s >>= 1) {
    if (o < s) {
      red[0][o] += red[0][o + s];
      red[1][o] += red[1][o + s];
      red[2][o] += red[2][o + s];
    }
    __syncthreads();
  }
  __shared__ float inv[3];
  if (o == 0) {
#pragma unroll
    for (int t = 0; t < 3; ++t)
      inv[t] = 1.f / fmaxf(sqrtf(red[t][0]), 1e-12f);
  }
  __syncthreads();
  out[i] = res * inv[0];
  out[(size_t)N * OUT + i] = br1 * inv[1];
  out[2 * (size_t)N * OUT + i] = u4 * inv[2];
}

// ---------------------------------------------------------------------------
extern "C" void kernel_launch(void* const* d_in, const int* in_sizes, int n_in,
                              void* d_out, int out_size, void* d_ws, size_t ws_size,
                              hipStream_t stream) {
  const float* feature  = (const float*)d_in[0];
  const float* A_stack  = (const float*)d_in[1];
  const float* encode   = (const float*)d_in[2];
  const float* W1       = (const float*)d_in[3];
  const float* b1       = (const float*)d_in[4];
  const float* W2       = (const float*)d_in[5];
  const float* b2       = (const float*)d_in[6];
  const float* weight_b = (const float*)d_in[7];
  const float* relint   = (const float*)d_in[8];
  const float* strength = (const float*)d_in[9];
  const float* struct_w = (const float*)d_in[10];

  char* ws = (char*)d_ws;
  _Float16* finalA = (_Float16*)ws; ws += (size_t)N * N * sizeof(_Float16);   // 33.5 MB
  _Float16* H1T    = (_Float16*)ws; ws += (size_t)OUT * N * sizeof(_Float16);
  _Float16* H2T    = (_Float16*)ws; ws += (size_t)OUT * N * sizeof(_Float16);
  float* H1 = (float*)ws; ws += (size_t)N * OUT * sizeof(float);
  float* U1 = (float*)ws; ws += (size_t)N * OUT * sizeof(float);
  float* U2 = (float*)ws; ws += (size_t)N * OUT * sizeof(float);
  float* U3 = (float*)ws; ws += (size_t)N * OUT * sizeof(float);
  float* G  = (float*)ws; ws += (size_t)N * OUT * sizeof(float);
  float* U4 = (float*)ws; ws += (size_t)N * OUT * sizeof(float);
  float* S1 = (float*)ws; ws += 7 * OUT * sizeof(float);
  float* S2 = (float*)ws; ws += 7 * OUT * sizeof(float);

  // dense path
  build_final<<<dim3(64, 64), 256, 0, stream>>>(A_stack, weight_b, relint,
                                                strength, finalA);
  xw_small<<<(N * OUT) / 256, 256, 0, stream>>>(feature, W1, NFEAT, H1, H1T);
  gemm_wmma<<<N / 128, 256, 0, stream>>>(finalA, H1T, b1, U1);
  xw_small<<<(N * OUT) / 256, 256, 0, stream>>>(U1, W2, OUT, nullptr, H2T);
  gemm_wmma<<<N / 128, 256, 0, stream>>>(finalA, H2T, b2, U2);

  // rank-7 struct path (struct_adj never materialized)
  etx<<<7 * OUT, 256, 0, stream>>>(encode, H1, S1);
  exs<<<(N * OUT) / 256, 256, 0, stream>>>(encode, struct_w, S1, b1, U3);
  xw_small<<<(N * OUT) / 256, 256, 0, stream>>>(U3, W2, OUT, G, nullptr);
  etx<<<7 * OUT, 256, 0, stream>>>(encode, G, S2);
  exs<<<(N * OUT) / 256, 256, 0, stream>>>(encode, struct_w, S2, b2, U4);

  // combine + normalize -> concatenated tuple output
  finalize<<<N, 64, 0, stream>>>(U1, U2, U4, (float*)d_out);
}
